// WeaklySelector_61675730370890
// MI455X (gfx1250) — compile-verified, matrix operand order checked by reading, not run on hardware
//
#include <hip/hip_runtime.h>
#include <hip/hip_bf16.h>
#include <cstddef>

// ---------------------------------------------------------------------------
// MI455X (gfx1250) implementation.
// Problem: B=32, S=4096, C=256, NC=200, NUM_SELECT=128.
//  logits = x @ W + b ; score(token) = max softmax prob = 1 / sum(exp(l - lmax))
//  per batch: top-128 tokens by score (stable desc) ; gather x rows.
// HBM-bound (128 MB x read vs 13.4 GFLOP) -> single pass over x, bf16 v_wmma
// GEMM with cross-tile double-buffered B fragments (loads overlap WMMA),
// online softmax-max in registers, LDS bitonic top-k, L2-hit gather.
// ---------------------------------------------------------------------------

typedef __attribute__((ext_vector_type(16))) __bf16 v16bf;
typedef __attribute__((ext_vector_type(8)))  float  v8f;

union Frag16 {
    v16bf v;
    uint4 q[2];
    unsigned u[8];
};

#if __has_builtin(__builtin_amdgcn_sched_barrier)
#define SCHED_FENCE() __builtin_amdgcn_sched_barrier(0)
#else
#define SCHED_FENCE()
#endif

__device__ __forceinline__ unsigned pack_bf16(float a, float b) {
    // round-to-nearest-even bf16, packed (a -> low, b -> high)
    unsigned ua = __float_as_uint(a), ub = __float_as_uint(b);
    unsigned ra = (ua + 0x7FFFu + ((ua >> 16) & 1u)) >> 16;
    unsigned rb = (ub + 0x7FFFu + ((ub >> 16) & 1u)) >> 16;
    return (ra & 0xFFFFu) | (rb << 16);
}

__device__ __forceinline__ float hmax16(float v) {
    // butterfly max within groups of 16 lanes (wave32: two independent halves)
    for (int m = 8; m >= 1; m >>= 1) v = fmaxf(v, __shfl_xor(v, m, 16));
    return v;
}
__device__ __forceinline__ float hsum16(float v) {
    for (int m = 8; m >= 1; m >>= 1) v += __shfl_xor(v, m, 16);
    return v;
}

// ---------------------------------------------------------------------------
// Kernel 1: W (256x200 f32, row-major [c][n]) -> Wt bf16 [n][k], n padded to
// 224 (rows 200..223 zeroed) so the WMMA pipeline runs 14 uniform N-tiles.
// Packed u32 (K even low, K odd high): a B fragment is two 16B loads/lane.
// ~112 KB -> L2 resident for all blocks.
// ---------------------------------------------------------------------------
__global__ void prep_w_kernel(const float* __restrict__ W,
                              unsigned* __restrict__ Wt) {
    unsigned q = blockIdx.x * blockDim.x + threadIdx.x;   // 224*128 = 28672
    if (q >= 224u * 128u) return;
    unsigned n  = q >> 7;          // padded output column 0..223
    unsigned kp = q & 127u;        // K pair index, K = 2*kp, 2*kp+1
    unsigned val = 0u;
    if (n < 200u) {
        float w0 = W[(2u * kp) * 200u + n];
        float w1 = W[(2u * kp + 1u) * 200u + n];
        val = pack_bf16(w0, w1);
    }
    Wt[n * 128u + kp] = val;
}

// ---- load one tile's 8 B fragments (16 x global_load_b128, L2-resident) ----
__device__ __forceinline__ void loadB(Frag16* __restrict__ Bf,
                                      const unsigned* __restrict__ Wt,
                                      unsigned n, unsigned halfsel) {
    const unsigned* wrow = Wt + n * 128u;
    #pragma unroll
    for (int ks = 0; ks < 8; ++ks) {
        const unsigned kb = (unsigned)ks * 16u + halfsel * 4u;
        Bf[ks].q[0] = *(const uint4*)(wrow + kb);
        Bf[ks].q[1] = *(const uint4*)(wrow + kb + 8u);
    }
}

// ---- one N-tile: 8 x v_wmma + online (max, sumexp) merge -------------------
__device__ __forceinline__ void tile_step(const Frag16* __restrict__ Af,
                                          const Frag16* __restrict__ Bf,
                                          unsigned n,
                                          const float* __restrict__ bias,
                                          float* __restrict__ m,
                                          float* __restrict__ Z) {
    v8f acc = {};
    #pragma unroll
    for (int ks = 0; ks < 8; ++ks) {
        acc = __builtin_amdgcn_wmma_f32_16x16x32_bf16(
            /*neg_a=*/false, Af[ks].v, /*neg_b=*/false, Bf[ks].v,
            /*c_mod=*/(short)0, acc, /*reuse_a=*/false, /*reuse_b=*/false);
    }
    const float nb = (n < 200u) ? bias[n] : 0.0f;
    #pragma unroll
    for (int j = 0; j < 8; ++j) {
        float v = acc[j] + nb;
        if (n >= 200u) v = -3.0e38f;                 // mask pad columns
        float mt = hmax16(v);
        float st = hsum16(__expf(v - mt));
        float mn = fmaxf(m[j], mt);
        // all-masked tiles: st*exp(mt-mn) = 16*exp(-3e38-m) == 0 -> no-op
        Z[j] = Z[j] * __expf(m[j] - mn) + st * __expf(mt - mn);
        m[j] = mn;
    }
}

// ---------------------------------------------------------------------------
// Kernel 2: score kernel.  1024 blocks x 256 threads (8 waves of 32).
// Each block: 128 tokens.  x tile staged in LDS as packed bf16; each wave
// loads its 16-token stripe ONCE into 8 A fragments (2 x ds_load_b128 each).
// N-tile loop is software-pipelined: B fragments for tile t+1 are issued
// (and fenced with sched_barrier) before the 8 v_wmma_f32_16x16x32_bf16 of
// tile t, so global loads overlap matrix math + softmax VALU instead of the
// load->wait(0)->wmma lock-step the scheduler otherwise produces.
// A-frag addressing (16-bit A 16x32, ISA 7.12.2):
//   lane l: M = l&15 ; Kbase = (l<16 ? 0 : 8) + 32*kstep
//   VGPR0-3: K = Kbase..Kbase+7 ; VGPR4-7: K = Kbase+16..+23
// C/D layout: lane = N (l&15), VGPR j -> M = (l<16 ? j : j+8).
// ---------------------------------------------------------------------------
__global__ __launch_bounds__(256) void score_kernel(
    const float* __restrict__ x,        // [131072][256]
    const unsigned* __restrict__ Wt,    // [224][128] packed bf16 pairs
    const float* __restrict__ bias,     // [200]
    float* __restrict__ scores)         // [131072]
{
    __shared__ unsigned Albs[128 * 128];   // 128 tokens x 256 bf16 = 64 KB

    const unsigned tid = threadIdx.x;
    const size_t tok0  = (size_t)blockIdx.x * 128u;

    // ---- stage x tile -> LDS packed bf16 (coalesced b128 loads) ----
    const float4* xin = (const float4*)(x + tok0 * 256u);
    #pragma unroll 4
    for (int i = 0; i < 32; ++i) {
        unsigned q = tid + (unsigned)i * 256u;     // float4 index, 0..8191
        float4 f = xin[q];
        Albs[2u * q]      = pack_bf16(f.x, f.y);
        Albs[2u * q + 1u] = pack_bf16(f.z, f.w);
    }
    __syncthreads();

    const unsigned w       = tid >> 5;     // wave id 0..7  (wave32)
    const unsigned l       = tid & 31u;    // lane
    const unsigned lm      = l & 15u;
    const unsigned halfsel = (l >> 4) & 1u;          // lanes 16-31 -> 1
    const unsigned arow    = (w * 16u + lm) * 128u;  // A row base (uints)

    // ---- load this wave's 16-token A stripe once: 8 fragments, 64 VGPRs ----
    Frag16 Af[8];
    #pragma unroll
    for (int ks = 0; ks < 8; ++ks) {
        const unsigned kb = (unsigned)ks * 16u + halfsel * 4u;
        Af[ks].q[0] = *(const uint4*)(&Albs[arow + kb]);
        Af[ks].q[1] = *(const uint4*)(&Albs[arow + kb + 8u]);
    }

    float m[8], Z[8];
    #pragma unroll
    for (int j = 0; j < 8; ++j) { m[j] = -3.0e38f; Z[j] = 0.0f; }

    // ---- software-pipelined N-tile loop (14 tiles, 2 per iteration) ----
    Frag16 B0[8], B1[8];
    loadB(B0, Wt, lm, halfsel);                          // tile 0
    for (int t = 0; t < 14; t += 2) {
        loadB(B1, Wt, (unsigned)(t + 1) * 16u + lm, halfsel);   // prefetch t+1
        SCHED_FENCE();
        tile_step(Af, B0, (unsigned)t * 16u + lm, bias, m, Z);  // compute t
        if (t + 2 < 14)
            loadB(B0, Wt, (unsigned)(t + 2) * 16u + lm, halfsel); // prefetch t+2
        SCHED_FENCE();
        tile_step(Af, B1, (unsigned)(t + 1) * 16u + lm, bias, m, Z); // compute t+1
    }

    // lane lm==0 of each half holds the reduced values (replicated in half)
    if (lm == 0u) {
        #pragma unroll
        for (int j = 0; j < 8; ++j) {
            unsigned token = w * 16u + (unsigned)j + halfsel * 8u;
            scores[tok0 + token] = 1.0f / Z[j];          // == max softmax prob
        }
    }
}

// ---------------------------------------------------------------------------
// Kernel 3: per-batch bitonic sort of 4096 (score, idx) keys, emit top 128.
// Key = (ordered(score) << 32) | (0xFFFFFFFF - idx): descending sort gives
// score-descending, index-ascending on ties == stable argsort(-score).
// ---------------------------------------------------------------------------
__global__ __launch_bounds__(1024) void topk_kernel(
    const float* __restrict__ scores,   // [32][4096]
    int* __restrict__ topidx)           // [32][128]
{
    __shared__ unsigned long long keys[4096];            // 32 KB
    const int b   = blockIdx.x;
    const int tid = threadIdx.x;

    for (int i = tid; i < 4096; i += 1024) {
        unsigned u = __float_as_uint(scores[b * 4096 + i]);
        u = (u & 0x80000000u) ? ~u : (u | 0x80000000u);  // order-preserving
        keys[i] = ((unsigned long long)u << 32) |
                  (unsigned long long)(0xFFFFFFFFu - (unsigned)i);
    }
    __syncthreads();

    for (int k = 2; k <= 4096; k <<= 1) {
        for (int j = k >> 1; j > 0; j >>= 1) {
            for (int i = tid; i < 4096; i += 1024) {
                int ixj = i ^ j;
                if (ixj > i) {
                    unsigned long long a = keys[i], c = keys[ixj];
                    bool desc = ((i & k) == 0);
                    if (desc ? (a < c) : (a > c)) { keys[i] = c; keys[ixj] = a; }
                }
            }
            __syncthreads();
        }
    }

    if (tid < 128) {
        topidx[b * 128 + tid] =
            (int)(0xFFFFFFFFu - (unsigned)(keys[tid] & 0xFFFFFFFFull));
    }
}

// ---------------------------------------------------------------------------
// Kernel 4: gather selected rows.  out[b][j][:] = x[b][top[b][j]][:]
// 4 MB, rows come out of L2 (x fits in the 192 MB global L2).
// ---------------------------------------------------------------------------
__global__ __launch_bounds__(64) void gather_kernel(
    const float* __restrict__ x,
    const int* __restrict__ topidx,
    float* __restrict__ out)
{
    const int b = blockIdx.x >> 7;
    const int j = blockIdx.x & 127;
    const int row = topidx[b * 128 + j];
    const float4* src = (const float4*)(x + ((size_t)b * 4096u + (unsigned)row) * 256u);
    float4*       dst = (float4*)(out + ((size_t)b * 128u + (unsigned)j) * 256u);
    dst[threadIdx.x] = src[threadIdx.x];
}

// ---------------------------------------------------------------------------
extern "C" void kernel_launch(void* const* d_in, const int* in_sizes, int n_in,
                              void* d_out, int out_size, void* d_ws, size_t ws_size,
                              hipStream_t stream) {
    const float* x = (const float*)d_in[0];   // [32][4096][256] f32
    const float* W = (const float*)d_in[1];   // [256][200] f32
    const float* b = (const float*)d_in[2];   // [200] f32
    float* out = (float*)d_out;               // [32][128][256] f32

    // workspace layout
    unsigned* wt_u32 = (unsigned*)d_ws;                           // 114,688 B
    float*    scores = (float*)((char*)d_ws + 114688);            // 524,288 B
    int*      topidx = (int*)((char*)d_ws + 114688 + 524288);     //  16,384 B

    // 1) W -> bf16 [n][k] transpose (pad n to 224)
    prep_w_kernel<<<112, 256, 0, stream>>>(W, wt_u32);
    // 2) logits GEMM (bf16 WMMA, double-buffered B) + online max-softmax score
    score_kernel<<<1024, 256, 0, stream>>>(x, wt_u32, b, scores);
    // 3) per-batch stable top-128 (bitonic)
    topk_kernel<<<32, 1024, 0, stream>>>(scores, topidx);
    // 4) gather selected rows
    gather_kernel<<<32 * 128, 64, 0, stream>>>(x, topidx, out);
}
